// BailingMoeV2SparseMoeBlock_51780125720958
// MI455X (gfx1250) — matrix operand order repaired and checked by reading, not compiled, not run on hardware
//
#include <hip/hip_runtime.h>

// ---------------------------------------------------------------------------
// BailingMoeV2 sparse MoE block for MI455X (gfx1250, wave32).
//   bf16 WMMA (v_wmma_f32_16x16x32_bf16) for all GEMMs, fp32 gate routing,
//   async global->LDS A-tile staging (global_load_async_to_lds_b128),
//   b128 weight streaming + v_cvt_pk_bf16_f32 conversion.
// ---------------------------------------------------------------------------

#define T_TOK   4096
#define H_DIM   2048
#define E_NUM   64
#define K_TOP   8
#define G_NUM   8
#define TOPKG   4
#define IE      512     // routed expert intermediate
#define IS      512     // shared expert intermediate
#define N2      1024    // 2*I (gate|up concatenated)
#define CAP     768
#define SCALE_F 2.5f
#define LSTR    40      // LDS row stride (bf16 elems): 80B pitch, 16B aligned

typedef __bf16 bf16_t;
typedef bf16_t v16bf __attribute__((ext_vector_type(16)));
typedef float  v8f   __attribute__((ext_vector_type(8)));
typedef float  v4f   __attribute__((ext_vector_type(4)));

union Frag { v16bf v; unsigned u[8]; };

__device__ inline unsigned short f2bf(float f) {
  return __builtin_bit_cast(unsigned short, (__bf16)f);
}
__device__ inline unsigned f2bf2(float lo, float hi) {   // pack 2 bf16
  return (unsigned)f2bf(lo) | ((unsigned)f2bf(hi) << 16);
}

__device__ inline v8f wmma_bf16(v16bf a, v16bf b, v8f c) {
  return __builtin_amdgcn_wmma_f32_16x16x32_bf16(
      false, a, false, b, (short)0, c, false, false);
}

// Async copy 16B global -> LDS (ASYNCcnt tracked).
__device__ inline void async_cp16(unsigned lds_off, const void* gsrc) {
  asm volatile("global_load_async_to_lds_b128 %0, %1, off"
               :: "v"(lds_off), "v"(gsrc) : "memory");
}
__device__ inline void async_wait0() {
  asm volatile("s_wait_asynccnt 0x0" ::: "memory");
}
__device__ inline unsigned lds_off32(const void* p) {    // low 32b = LDS offset
  return (unsigned)(unsigned long long)p;
}

// A fragment: 16x32 bf16; lanes 0-15/16-31 both M=0..15; VGPR q holds
// K pairs {2q | 2q+8} (+8*sel) per ISA table.
__device__ inline v16bf load_fragA(const unsigned short* lds, int row0, int lane) {
  int sel = (lane >> 4) & 1;
  const unsigned short* p = lds + (row0 + (lane & 15)) * LSTR;
  Frag f;
#pragma unroll
  for (int q = 0; q < 8; ++q) {
    int k = ((q < 4) ? (2 * q) : (2 * q + 8)) + 8 * sel;
    f.u[q] = *(const unsigned*)(p + k);
  }
  return f.v;
}

// B fragment: 32x16 bf16; VGPR q holds K pairs {2q | 2q+8} (+8*sel), same as A.
// LDS tile is stored transposed (n-major, k-contiguous).
__device__ inline v16bf load_fragB(const unsigned short* lds, int col0, int lane) {
  int sel = (lane >> 4) & 1;
  const unsigned short* p = lds + (col0 + (lane & 15)) * LSTR;
  Frag f;
#pragma unroll
  for (int q = 0; q < 8; ++q) {
    int k = ((q < 4) ? (2 * q) : (2 * q + 8)) + 8 * sel;
    f.u[q] = *(const unsigned*)(p + k);
  }
  return f.v;
}

__device__ inline float silu_f(float g) { return g / (1.f + __expf(-g)); }

// ---------------------------------------------------------------------------
__global__ void k_init(int* counts, unsigned* zrow) {
  if (threadIdx.x < E_NUM) counts[threadIdx.x] = 0;
  for (int i = threadIdx.x; i < H_DIM / 2; i += 256) zrow[i] = 0u;
}

__global__ void k_cast_x(const float* __restrict__ x, unsigned short* __restrict__ xb) {
  int i = blockIdx.x * blockDim.x + threadIdx.x;        // one float4 each
  if (i < T_TOK * H_DIM / 4) {
    v4f r = *(const v4f*)(x + 4 * (size_t)i);
    unsigned lo = f2bf2(r[0], r[1]);
    unsigned hi = f2bf2(r[2], r[3]);
    *(unsigned*)(xb + 4 * (size_t)i)     = lo;
    *(unsigned*)(xb + 4 * (size_t)i + 2) = hi;
  }
}

// fp32 gate GEMM + sigmoid + grouped top-k routing. One block (64 thr)/token.
__global__ void k_gate(const float* __restrict__ x, const float* __restrict__ gw,
                       const float* __restrict__ bias,
                       int* __restrict__ topk_idx, float* __restrict__ topk_w) {
  __shared__ float sc[E_NUM];
  __shared__ float sr[E_NUM];
  int t = blockIdx.x, e = threadIdx.x;
  const float* xr = x + (size_t)t * H_DIM;
  float acc = 0.f;
  for (int h = 0; h < H_DIM; ++h) acc += xr[h] * gw[(size_t)h * E_NUM + e];
  float s = 1.f / (1.f + __expf(-acc));
  sc[e] = s;
  sr[e] = s + bias[e];
  __syncthreads();
  if (e == 0) {
    float gs[G_NUM];
    for (int g = 0; g < G_NUM; ++g) {                 // sum of top-2 per group
      float m1 = -1e30f, m2 = -1e30f;
      for (int j = 0; j < E_NUM / G_NUM; ++j) {
        float v = sr[g * (E_NUM / G_NUM) + j];
        if (v > m1) { m2 = m1; m1 = v; } else if (v > m2) { m2 = v; }
      }
      gs[g] = m1 + m2;
    }
    bool gsel[G_NUM];
    for (int g = 0; g < G_NUM; ++g) gsel[g] = false;
    for (int it = 0; it < TOPKG; ++it) {              // top-4 groups, ties->low
      float best = -1e30f; int bi = 0;
      for (int g = 0; g < G_NUM; ++g)
        if (!gsel[g] && gs[g] > best) { best = gs[g]; bi = g; }
      gsel[bi] = true;
    }
    bool esel[E_NUM];
    for (int q = 0; q < E_NUM; ++q) esel[q] = false;
    int idx[K_TOP]; float wv[K_TOP]; float wsum = 0.f;
    for (int k = 0; k < K_TOP; ++k) {                 // top-8 masked experts
      float best = -1e30f; int bi = 0;
      for (int q = 0; q < E_NUM; ++q)
        if (gsel[q >> 3] && !esel[q] && sr[q] > best) { best = sr[q]; bi = q; }
      esel[bi] = true; idx[k] = bi; wv[k] = sc[bi]; wsum += sc[bi];
    }
    float inv = SCALE_F / (wsum + 1e-20f);
    for (int k = 0; k < K_TOP; ++k) {
      topk_idx[t * K_TOP + k] = idx[k];
      topk_w[t * K_TOP + k]   = wv[k] * inv;
    }
  }
}

// Bucket (token, weight) per expert; slot order irrelevant (summed output).
__global__ void k_build(const int* __restrict__ topk_idx, const float* __restrict__ topk_w,
                        int* __restrict__ counts, int* __restrict__ tok_list,
                        float* __restrict__ tok_w) {
  int p = blockIdx.x * blockDim.x + threadIdx.x;
  if (p >= T_TOK * K_TOP) return;
  int e = topk_idx[p];
  int pos = atomicAdd(&counts[e], 1);
  if (pos < CAP) {
    tok_list[e * CAP + pos] = p / K_TOP;
    tok_w[e * CAP + pos]    = topk_w[p];
  }
}

// -------- shared expert: gate_up GEMM + silu*u -> bf16 activations ---------
__global__ void __launch_bounds__(256)
k_shared_gateup(const unsigned short* __restrict__ xb, const float* __restrict__ wgu,
                unsigned short* __restrict__ aout) {
  __shared__ unsigned short As[64 * LSTR];
  __shared__ unsigned short Bg[64 * LSTR];
  __shared__ unsigned short Bu[64 * LSTR];
  int t0 = blockIdx.x * 64, n0 = blockIdx.y * 64;
  int tid = threadIdx.x, lane = tid & 31, wv = tid >> 5;
  int mr = wv & 3, nrb = (wv >> 2) * 2;
  int ar = tid >> 2, ac = (tid & 3) * 8;              // A: 16B chunk per thread
  const unsigned short* abase = xb + (size_t)(t0 + ar) * H_DIM + ac;
  unsigned aoff = lds_off32(As + ar * LSTR + ac);
  int bk = (tid >> 4) * 2, bn = (tid & 15) * 4;       // B: k-pair x n-quad unit
  v8f accG[2] = {{0,0,0,0,0,0,0,0},{0,0,0,0,0,0,0,0}};
  v8f accU[2] = {{0,0,0,0,0,0,0,0},{0,0,0,0,0,0,0,0}};
  for (int k0 = 0; k0 < H_DIM; k0 += 32) {
    async_cp16(aoff, abase + k0);                     // A tile: async -> LDS
    {
      const float* wp = wgu + (size_t)(k0 + bk) * N2 + n0 + bn;
      v4f g0 = *(const v4f*)(wp);
      v4f g1 = *(const v4f*)(wp + N2);
      v4f u0 = *(const v4f*)(wp + IS);
      v4f u1 = *(const v4f*)(wp + N2 + IS);
      if (k0 + 32 < H_DIM) __builtin_prefetch(wp + 32 * N2, 0, 1);
#pragma unroll
      for (int jj = 0; jj < 4; ++jj) {
        *(unsigned*)(Bg + (bn + jj) * LSTR + bk) = f2bf2(g0[jj], g1[jj]);
        *(unsigned*)(Bu + (bn + jj) * LSTR + bk) = f2bf2(u0[jj], u1[jj]);
      }
    }
    async_wait0();
    __syncthreads();
    v16bf a = load_fragA(As, mr * 16, lane);
#pragma unroll
    for (int j = 0; j < 2; ++j) {
      v16bf bg = load_fragB(Bg, (nrb + j) * 16, lane);
      accG[j] = wmma_bf16(a, bg, accG[j]);
      v16bf bu = load_fragB(Bu, (nrb + j) * 16, lane);
      accU[j] = wmma_bf16(a, bu, accU[j]);
    }
    __syncthreads();
  }
  int sel = (lane >> 4) & 1;
#pragma unroll
  for (int j = 0; j < 2; ++j)
#pragma unroll
    for (int v = 0; v < 8; ++v) {
      int m = mr * 16 + v + 8 * sel;
      int n = (nrb + j) * 16 + (lane & 15);
      aout[(size_t)(t0 + m) * IS + n0 + n] = f2bf(silu_f(accG[j][v]) * accU[j][v]);
    }
}

// -------- shared expert down GEMM: writes out (initializes all of it) ------
__global__ void __launch_bounds__(256)
k_shared_down(const unsigned short* __restrict__ ash, const float* __restrict__ wd,
              float* __restrict__ out) {
  __shared__ unsigned short As[64 * LSTR];
  __shared__ unsigned short Bs[64 * LSTR];
  int t0 = blockIdx.x * 64, n0 = blockIdx.y * 64;
  int tid = threadIdx.x, lane = tid & 31, wv = tid >> 5;
  int mr = wv & 3, nrb = (wv >> 2) * 2;
  int ar = tid >> 2, ac = (tid & 3) * 8;
  const unsigned short* abase = ash + (size_t)(t0 + ar) * IS + ac;
  unsigned aoff = lds_off32(As + ar * LSTR + ac);
  int bk = (tid >> 4) * 2, bn = (tid & 15) * 4;
  v8f acc[2] = {{0,0,0,0,0,0,0,0},{0,0,0,0,0,0,0,0}};
  for (int k0 = 0; k0 < IS; k0 += 32) {
    async_cp16(aoff, abase + k0);
    {
      const float* wp = wd + (size_t)(k0 + bk) * H_DIM + n0 + bn;
      v4f r0 = *(const v4f*)(wp);
      v4f r1 = *(const v4f*)(wp + H_DIM);
#pragma unroll
      for (int jj = 0; jj < 4; ++jj)
        *(unsigned*)(Bs + (bn + jj) * LSTR + bk) = f2bf2(r0[jj], r1[jj]);
    }
    async_wait0();
    __syncthreads();
    v16bf a = load_fragA(As, mr * 16, lane);
#pragma unroll
    for (int j = 0; j < 2; ++j) {
      v16bf b = load_fragB(Bs, (nrb + j) * 16, lane);
      acc[j] = wmma_bf16(a, b, acc[j]);
    }
    __syncthreads();
  }
  int sel = (lane >> 4) & 1;
#pragma unroll
  for (int j = 0; j < 2; ++j)
#pragma unroll
    for (int v = 0; v < 8; ++v) {
      int m = mr * 16 + v + 8 * sel;
      int n = (nrb + j) * 16 + (lane & 15);
      out[(size_t)(t0 + m) * H_DIM + n0 + n] = acc[j][v];
    }
}

// -------- routed experts: gate_up on gathered rows + silu*u ----------------
__global__ void __launch_bounds__(256)
k_exp_gateup(const unsigned short* __restrict__ xb, const float* __restrict__ w_gate_up,
             const int* __restrict__ counts, const int* __restrict__ tok_list,
             const unsigned short* __restrict__ zrow, unsigned short* __restrict__ aout) {
  __shared__ unsigned short As[64 * LSTR];
  __shared__ unsigned short Bg[64 * LSTR];
  __shared__ unsigned short Bu[64 * LSTR];
  __shared__ int rowTok[64];
  int e = blockIdx.z;
  int cnt = counts[e]; if (cnt > CAP) cnt = CAP;
  int m0 = blockIdx.x * 64;
  if (m0 >= cnt) return;                              // block-uniform exit
  int n0 = blockIdx.y * 64;
  int tid = threadIdx.x, lane = tid & 31, wv = tid >> 5;
  int mr = wv & 3, nrb = (wv >> 2) * 2;
  const float* wgu = w_gate_up + (size_t)e * H_DIM * N2;
  if (tid < 64) rowTok[tid] = (m0 + tid < cnt) ? tok_list[e * CAP + m0 + tid] : -1;
  __syncthreads();
  int ar = tid >> 2, ac = (tid & 3) * 8;
  int tkn = rowTok[ar];                               // padded rows -> zero row
  const unsigned short* abase =
      (tkn >= 0 ? xb + (size_t)tkn * H_DIM : zrow) + ac;
  unsigned aoff = lds_off32(As + ar * LSTR + ac);
  int bk = (tid >> 4) * 2, bn = (tid & 15) * 4;
  v8f accG[2] = {{0,0,0,0,0,0,0,0},{0,0,0,0,0,0,0,0}};
  v8f accU[2] = {{0,0,0,0,0,0,0,0},{0,0,0,0,0,0,0,0}};
  for (int k0 = 0; k0 < H_DIM; k0 += 32) {
    async_cp16(aoff, abase + k0);
    {
      const float* wp = wgu + (size_t)(k0 + bk) * N2 + n0 + bn;
      v4f g0 = *(const v4f*)(wp);
      v4f g1 = *(const v4f*)(wp + N2);
      v4f u0 = *(const v4f*)(wp + IE);
      v4f u1 = *(const v4f*)(wp + N2 + IE);
      if (k0 + 32 < H_DIM) __builtin_prefetch(wp + 32 * N2, 0, 1);
#pragma unroll
      for (int jj = 0; jj < 4; ++jj) {
        *(unsigned*)(Bg + (bn + jj) * LSTR + bk) = f2bf2(g0[jj], g1[jj]);
        *(unsigned*)(Bu + (bn + jj) * LSTR + bk) = f2bf2(u0[jj], u1[jj]);
      }
    }
    async_wait0();
    __syncthreads();
    v16bf a = load_fragA(As, mr * 16, lane);
#pragma unroll
    for (int j = 0; j < 2; ++j) {
      v16bf bg = load_fragB(Bg, (nrb + j) * 16, lane);
      accG[j] = wmma_bf16(a, bg, accG[j]);
      v16bf bu = load_fragB(Bu, (nrb + j) * 16, lane);
      accU[j] = wmma_bf16(a, bu, accU[j]);
    }
    __syncthreads();
  }
  int sel = (lane >> 4) & 1;
#pragma unroll
  for (int j = 0; j < 2; ++j)
#pragma unroll
    for (int v = 0; v < 8; ++v) {
      int m = mr * 16 + v + 8 * sel;
      int n = (nrb + j) * 16 + (lane & 15);
      aout[((size_t)e * CAP + m0 + m) * IE + n0 + n] =
          f2bf(silu_f(accG[j][v]) * accU[j][v]);      // zero rows stay 0
    }
}

// -------- routed experts: down GEMM + weighted atomic scatter-add ----------
__global__ void __launch_bounds__(256)
k_exp_down(const unsigned short* __restrict__ aexp, const float* __restrict__ w_down,
           const int* __restrict__ counts, const int* __restrict__ tok_list,
           const float* __restrict__ tok_w, float* __restrict__ out) {
  __shared__ unsigned short As[64 * LSTR];
  __shared__ unsigned short Bs[64 * LSTR];
  int e = blockIdx.z;
  int cnt = counts[e]; if (cnt > CAP) cnt = CAP;
  int m0 = blockIdx.x * 64;
  if (m0 >= cnt) return;
  int n0 = blockIdx.y * 64;
  int tid = threadIdx.x, lane = tid & 31, wv = tid >> 5;
  int mr = wv & 3, nrb = (wv >> 2) * 2;
  const unsigned short* ae = aexp + (size_t)e * CAP * IE;
  const float* wd = w_down + (size_t)e * IE * H_DIM;
  int ar = tid >> 2, ac = (tid & 3) * 8;
  const unsigned short* abase = ae + (size_t)(m0 + ar) * IE + ac;
  unsigned aoff = lds_off32(As + ar * LSTR + ac);
  int bk = (tid >> 4) * 2, bn = (tid & 15) * 4;
  v8f acc[2] = {{0,0,0,0,0,0,0,0},{0,0,0,0,0,0,0,0}};
  for (int k0 = 0; k0 < IE; k0 += 32) {
    async_cp16(aoff, abase + k0);
    {
      const float* wp = wd + (size_t)(k0 + bk) * H_DIM + n0 + bn;
      v4f r0 = *(const v4f*)(wp);
      v4f r1 = *(const v4f*)(wp + H_DIM);
      if (k0 + 32 < IE) __builtin_prefetch(wp + 32 * H_DIM, 0, 1);
#pragma unroll
      for (int jj = 0; jj < 4; ++jj)
        *(unsigned*)(Bs + (bn + jj) * LSTR + bk) = f2bf2(r0[jj], r1[jj]);
    }
    async_wait0();
    __syncthreads();
    v16bf a = load_fragA(As, mr * 16, lane);
#pragma unroll
    for (int j = 0; j < 2; ++j) {
      v16bf b = load_fragB(Bs, (nrb + j) * 16, lane);
      acc[j] = wmma_bf16(a, b, acc[j]);
    }
    __syncthreads();
  }
  int sel = (lane >> 4) & 1;
#pragma unroll
  for (int j = 0; j < 2; ++j)
#pragma unroll
    for (int v = 0; v < 8; ++v) {
      int m = mr * 16 + v + 8 * sel;
      int gm = m0 + m;
      if (gm < cnt) {
        int   t = tok_list[e * CAP + gm];
        float w = tok_w[e * CAP + gm];
        int   n = (nrb + j) * 16 + (lane & 15);
        atomicAdd(&out[(size_t)t * H_DIM + n0 + n], w * acc[j][v]);
      }
    }
}

// ---------------------------------------------------------------------------
extern "C" void kernel_launch(void* const* d_in, const int* in_sizes, int n_in,
                              void* d_out, int out_size, void* d_ws, size_t ws_size,
                              hipStream_t stream) {
  const float* x           = (const float*)d_in[0];
  // d_in[1]=image_mask, d_in[2]=audio_mask: unused by reference math
  const float* gate_w      = (const float*)d_in[3];
  const float* expert_bias = (const float*)d_in[4];
  const float* w_gate_up   = (const float*)d_in[5];
  const float* w_down      = (const float*)d_in[6];
  const float* sh_wgu      = (const float*)d_in[7];
  const float* sh_wd       = (const float*)d_in[8];
  float* out = (float*)d_out;

  char* p = (char*)d_ws;
  auto take = [&](size_t bytes) -> char* {
    char* r = p; p += (bytes + 255) & ~(size_t)255; return r;
  };
  unsigned short* xb   = (unsigned short*)take((size_t)T_TOK * H_DIM * 2);
  unsigned short* a_sh = (unsigned short*)take((size_t)T_TOK * IS * 2);
  unsigned short* a_ex = (unsigned short*)take((size_t)E_NUM * CAP * IE * 2);
  unsigned short* zrow = (unsigned short*)take((size_t)H_DIM * 2);
  int*   topk_idx = (int*)  take((size_t)T_TOK * K_TOP * 4);
  float* topk_w   = (float*)take((size_t)T_TOK * K_TOP * 4);
  int*   counts   = (int*)  take(E_NUM * 4);
  int*   tok_list = (int*)  take((size_t)E_NUM * CAP * 4);
  float* tok_w    = (float*)take((size_t)E_NUM * CAP * 4);
  (void)in_sizes; (void)n_in; (void)out_size; (void)ws_size;

  k_init<<<1, 256, 0, stream>>>(counts, (unsigned*)zrow);
  k_cast_x<<<(T_TOK * H_DIM / 4 + 255) / 256, 256, 0, stream>>>(x, xb);
  k_gate<<<T_TOK, 64, 0, stream>>>(x, gate_w, expert_bias, topk_idx, topk_w);
  k_build<<<(T_TOK * K_TOP) / 256, 256, 0, stream>>>(topk_idx, topk_w, counts,
                                                     tok_list, tok_w);
  k_shared_gateup<<<dim3(T_TOK / 64, IS / 64), 256, 0, stream>>>(xb, sh_wgu, a_sh);
  k_shared_down  <<<dim3(T_TOK / 64, H_DIM / 64), 256, 0, stream>>>(a_sh, sh_wd, out);
  k_exp_gateup<<<dim3(CAP / 64, IE / 64, E_NUM), 256, 0, stream>>>(
      xb, w_gate_up, counts, tok_list, zrow, a_ex);
  k_exp_down<<<dim3(CAP / 64, H_DIM / 64, E_NUM), 256, 0, stream>>>(
      a_ex, w_down, counts, tok_list, tok_w, out);
}
